// Method_GCN_841813590223
// MI455X (gfx1250) — matrix operand order repaired and checked by reading, not compiled, not run on hardware
//
#include <hip/hip_runtime.h>
#include <hip/hip_bf16.h>

typedef __attribute__((ext_vector_type(16))) __bf16 v16bf;
typedef __attribute__((ext_vector_type(8)))  __bf16 v8bf;
typedef __attribute__((ext_vector_type(4)))  __bf16 v4bf;
typedef __attribute__((ext_vector_type(8)))  float  v8f;
typedef __attribute__((ext_vector_type(4)))  float  v4f;

#define F_IN 512
#define HID  128
#define CLS  40
#define CLSP 64

// Guaranteed native fp32 atomic (atomic-without-return -> STOREcnt; the
// implicit wait-idle at s_endpgm + stream ordering give cross-kernel
// visibility through L2 at device scope).
static __device__ __forceinline__ void atomAdd(float* p, float v) {
    asm volatile("global_atomic_add_f32 %0, %1, off"
                 :: "v"(p), "v"(v) : "memory");
}

// ---------------- degree / normalization ----------------
__global__ void k_deg_init(float* deg, int n) {
    int i = blockIdx.x * blockDim.x + threadIdx.x;
    if (i < n) deg[i] = 1.0f;                       // self loop
}
__global__ void k_deg(const int* __restrict__ ei, float* deg, int E) {
    int i = blockIdx.x * blockDim.x + threadIdx.x;
    if (i < E) atomAdd(&deg[ei[E + i]], 1.0f);      // dst row
}
__global__ void k_dinv(const float* __restrict__ deg, float* dinv, int n) {
    int i = blockIdx.x * blockDim.x + threadIdx.x;
    if (i < n) dinv[i] = rsqrtf(deg[i]);            // deg >= 1 always
}

// ---------------- weight conversion (bf16, transposed [N][K]) ----------------
__global__ void k_cvtW1(const float* __restrict__ W1, __bf16* __restrict__ WT) {
    int t = blockIdx.x * blockDim.x + threadIdx.x;  // HID*F_IN
    if (t < HID * F_IN) {
        int nn = t >> 9, k = t & 511;
        WT[t] = (__bf16)W1[k * HID + nn];
    }
}
__global__ void k_cvtW2(const float* __restrict__ W2, __bf16* __restrict__ WT) {
    int t = blockIdx.x * blockDim.x + threadIdx.x;  // CLSP*HID
    if (t < CLSP * HID) {
        int nn = t >> 7, k = t & 127;
        WT[t] = (__bf16)(nn < CLS ? W2[k * CLS + nn] : 0.0f);
    }
}

// ---------------- GEMM1: h1 = x @ W1 (bf16 WMMA, f32 accum) ----------------
// block = 256 thr = 8 waves; one 16-row M panel, each wave one 16-col N tile.
// Epilogue also seeds a1 = dinv^2 * h1 (self-loop term), saving a full pass.
__global__ __launch_bounds__(256) void k_gemm1(const float* __restrict__ x,
                                               const __bf16* __restrict__ WT,
                                               const float* __restrict__ dinv,
                                               float* __restrict__ h1,
                                               float* __restrict__ a1) {
    __shared__ __align__(16) __bf16 As[16][520];    // 512 + 8 pad (bank shift)
    const int m0 = blockIdx.x * 16;
    const int t  = threadIdx.x;
    {   // stage x panel -> bf16 LDS, coalesced float4 loads
        const int row = t >> 4, fc = t & 15;
        const float* src = x + (size_t)(m0 + row) * F_IN;
#pragma unroll
        for (int i = 0; i < 8; ++i) {
            int k4 = fc + 16 * i;                   // float4 index 0..127
            v4f v = *(const v4f*)(src + 4 * k4);
            v4bf b;
            b[0]=(__bf16)v[0]; b[1]=(__bf16)v[1]; b[2]=(__bf16)v[2]; b[3]=(__bf16)v[3];
            *(v4bf*)&As[row][4 * k4] = b;
        }
    }
    __syncthreads();
    const int w = t >> 5, lane = t & 31;
    const int hrow   = lane & 15;
    const int khalfA = (lane >> 4) * 8;             // A: K {0-7,16-23}/{8-15,24-31}
    const int kbB    = (lane >> 4) * 8;             // B: K {0-7,16-23}/{8-15,24-31}
    const __bf16* bptr = WT + (size_t)(w * 16 + hrow) * F_IN;
    v8f acc = {};
#pragma unroll
    for (int kc = 0; kc < F_IN; kc += 32) {
        v8bf alo = *(const v8bf*)&As[hrow][kc + khalfA];
        v8bf ahi = *(const v8bf*)&As[hrow][kc + khalfA + 16];
        v16bf a = __builtin_shufflevector(alo, ahi, 0,1,2,3,4,5,6,7,8,9,10,11,12,13,14,15);
        v8bf blo = *(const v8bf*)(bptr + kc + kbB);
        v8bf bhi = *(const v8bf*)(bptr + kc + kbB + 16);
        v16bf b = __builtin_shufflevector(blo, bhi, 0,1,2,3,4,5,6,7,8,9,10,11,12,13,14,15);
        acc = __builtin_amdgcn_wmma_f32_16x16x32_bf16(false, a, false, b,
                                                      (short)0, acc, false, false);
    }
    const int col   = w * 16 + hrow;
    const int mbase = m0 + (lane >> 4) * 8;         // D: VGPR g -> M = g (+8 hi lanes)
#pragma unroll
    for (int g = 0; g < 8; ++g) {
        const size_t idx = (size_t)(mbase + g) * HID + col;
        const float hv = acc[g];
        float s = dinv[mbase + g];
        h1[idx] = hv;
        a1[idx] = s * s * hv;                       // self-loop init of aggregation
    }
}

// ---------------- layer-1 edge scatter (wave per edge) ----------------
__global__ void k_edge1(const int* __restrict__ ei, const float* __restrict__ dinv,
                        const float* __restrict__ h1, float* __restrict__ a1, int E) {
    int t = blockIdx.x * blockDim.x + threadIdx.x;  // lane = 4 feats
    int e = t >> 5, lane = t & 31;
    if (e < E) {
        int s = ei[e], d = ei[E + e];
        float nrm = dinv[s] * dinv[d];
        v4f v = *(const v4f*)(h1 + (size_t)s * HID + 4 * lane);
        float* dp = a1 + (size_t)d * HID + 4 * lane;
        atomAdd(dp + 0, nrm * v[0]); atomAdd(dp + 1, nrm * v[1]);
        atomAdd(dp + 2, nrm * v[2]); atomAdd(dp + 3, nrm * v[3]);
    }
}

// ---------------- GEMM2: h2 = relu(a1+b1) @ W2 ----------------
// block = 256 thr = 8 waves = 2 M-tiles x 4 N-tiles (N padded 64, stores masked
// to 40). Bias+ReLU fused into LDS staging; epilogue seeds out = b2 + dinv^2*h2.
__global__ __launch_bounds__(256) void k_gemm2(const float* __restrict__ a1,
                                               const float* __restrict__ b1,
                                               const __bf16* __restrict__ WT,
                                               const float* __restrict__ dinv,
                                               const float* __restrict__ b2,
                                               float* __restrict__ h2,
                                               float* __restrict__ out) {
    __shared__ __align__(16) __bf16 As[32][136];    // 128 + 8 pad
    const int m0 = blockIdx.x * 32;
    const int t  = threadIdx.x;
    {   // stage relu(a1 + b1) -> bf16 LDS
        const int row = t >> 3, fc = t & 7;
        const float* src = a1 + (size_t)(m0 + row) * HID;
#pragma unroll
        for (int i = 0; i < 4; ++i) {
            int k4 = fc + 8 * i;                    // float4 index 0..31
            v4f v  = *(const v4f*)(src + 4 * k4);
            v4f bb = *(const v4f*)(b1 + 4 * k4);
            v4bf o;
#pragma unroll
            for (int j = 0; j < 4; ++j) {
                float r = v[j] + bb[j];
                o[j] = (__bf16)(r > 0.0f ? r : 0.0f);
            }
            *(v4bf*)&As[row][4 * k4] = o;
        }
    }
    __syncthreads();
    const int w = t >> 5, lane = t & 31;
    const int mt = w >> 2, nt = w & 3;
    const int hrow = lane & 15, khalfA = (lane >> 4) * 8, kbB = (lane >> 4) * 8;
    const __bf16* bptr = WT + (size_t)(nt * 16 + hrow) * HID;
    v8f acc = {};
#pragma unroll
    for (int kc = 0; kc < HID; kc += 32) {
        v8bf alo = *(const v8bf*)&As[mt * 16 + hrow][kc + khalfA];
        v8bf ahi = *(const v8bf*)&As[mt * 16 + hrow][kc + khalfA + 16];
        v16bf a = __builtin_shufflevector(alo, ahi, 0,1,2,3,4,5,6,7,8,9,10,11,12,13,14,15);
        v8bf blo = *(const v8bf*)(bptr + kc + kbB);
        v8bf bhi = *(const v8bf*)(bptr + kc + kbB + 16);
        v16bf b = __builtin_shufflevector(blo, bhi, 0,1,2,3,4,5,6,7,8,9,10,11,12,13,14,15);
        acc = __builtin_amdgcn_wmma_f32_16x16x32_bf16(false, a, false, b,
                                                      (short)0, acc, false, false);
    }
    const int col = nt * 16 + hrow;
    if (col < CLS) {
        const float bv = b2[col];
        const int mbase = m0 + mt * 16 + (lane >> 4) * 8;
#pragma unroll
        for (int g = 0; g < 8; ++g) {
            const size_t idx = (size_t)(mbase + g) * CLS + col;
            const float hv = acc[g];
            float s = dinv[mbase + g];
            h2[idx]  = hv;
            out[idx] = bv + s * s * hv;             // bias + self-loop init
        }
    }
}

// ---------------- layer-2 edge scatter ----------------
__global__ void k_edge2(const int* __restrict__ ei, const float* __restrict__ dinv,
                        const float* __restrict__ h2, float* __restrict__ out, int E) {
    int t = blockIdx.x * blockDim.x + threadIdx.x;  // E*64 slots, 40 active
    int e = t >> 6, f = t & 63;
    if (e < E && f < CLS) {
        int s = ei[e], d = ei[E + e];
        float nrm = dinv[s] * dinv[d];
        atomAdd(&out[(size_t)d * CLS + f], nrm * h2[(size_t)s * CLS + f]);
    }
}

extern "C" void kernel_launch(void* const* d_in, const int* in_sizes, int n_in,
                              void* d_out, int out_size, void* d_ws, size_t ws_size,
                              hipStream_t stream) {
    const float* x  = (const float*)d_in[0];
    const int*   ei = (const int*)  d_in[1];
    const float* W1 = (const float*)d_in[2];
    const float* b1 = (const float*)d_in[3];
    const float* W2 = (const float*)d_in[4];
    const float* b2 = (const float*)d_in[5];
    float* out = (float*)d_out;

    const int n = in_sizes[0] / F_IN;       // 100000 (multiple of 32)
    const int E = in_sizes[1] / 2;          // 1.6M

    // workspace carve-up (all fully rewritten every call; ~120 MB)
    float*  deg   = (float*)d_ws;
    float*  dinv  = deg  + n;
    float*  h1    = dinv + n;
    float*  a1    = h1 + (size_t)n * HID;
    float*  h2    = a1 + (size_t)n * HID;
    __bf16* W1bfT = (__bf16*)(h2 + (size_t)n * CLS);
    __bf16* W2bfT = W1bfT + (size_t)HID * F_IN;

    const int B = 256;
    auto blk = [](size_t work, int b) { return (unsigned)((work + b - 1) / b); };

    k_deg_init<<<blk(n, B), B, 0, stream>>>(deg, n);
    k_deg     <<<blk(E, B), B, 0, stream>>>(ei, deg, E);
    k_dinv    <<<blk(n, B), B, 0, stream>>>(deg, dinv, n);

    k_cvtW1<<<blk((size_t)HID * F_IN, B), B, 0, stream>>>(W1, W1bfT);
    k_cvtW2<<<blk((size_t)CLSP * HID, B), B, 0, stream>>>(W2, W2bfT);

    k_gemm1<<<(unsigned)(n / 16), B, 0, stream>>>(x, W1bfT, dinv, h1, a1);
    k_edge1<<<blk((size_t)E * 32, B), B, 0, stream>>>(ei, dinv, h1, a1, E);

    k_gemm2<<<(unsigned)(n / 32), B, 0, stream>>>(a1, b1, W2bfT, dinv, b2, h2, out);
    k_edge2<<<blk((size_t)E * 64, B), B, 0, stream>>>(ei, dinv, h2, out, E);
}
